// TripartiteHeteroEncoder_83270825935264
// MI455X (gfx1250) — matrix operand order repaired and checked by reading, not compiled, not run on hardware
//
#include <hip/hip_runtime.h>

#define HID 128

typedef __attribute__((ext_vector_type(16))) _Float16 v16h;
typedef __attribute__((ext_vector_type(8)))  _Float16 half8;
typedef __attribute__((ext_vector_type(8)))  float    v8f;
typedef __attribute__((ext_vector_type(4)))  float    v4f;

// ---------------------------------------------------------------------------
// Convert a 128x128 f32 weight (row-major [din][dout]) into transposed f16
// [dout][din] so WMMA B-fragments are contiguous 16-byte loads.
// ---------------------------------------------------------------------------
__global__ __launch_bounds__(256)
void cvt_tr_kernel(const float* __restrict__ W, _Float16* __restrict__ Wt) {
    int t = blockIdx.x * 256 + threadIdx.x;     // 16384 threads
    int j = t >> 7;          // output column of original W
    int i = t & 127;         // input row of original W
    Wt[j * 128 + i] = (_Float16)W[i * 128 + j];
}

// ---------------------------------------------------------------------------
// Encoder layer 0: h[n][j] = relu(x[n] * W0[j] + b0[j])   (W0 is 1x128)
// ---------------------------------------------------------------------------
__global__ __launch_bounds__(256)
void encode0_kernel(const float* __restrict__ x, const float* __restrict__ W0,
                    const float* __restrict__ b0, float* __restrict__ h) {
    long t = (long)blockIdx.x * 256 + threadIdx.x;
    int  j = (int)(t & 127);
    long i = t >> 7;
    float v = x[i] * W0[j] + b0[j];
    h[t] = v > 0.f ? v : 0.f;
}

// ---------------------------------------------------------------------------
// Edge gather->scale->scatter-add. 32 threads per edge, 4 feats per thread.
// ---------------------------------------------------------------------------
__global__ __launch_bounds__(256)
void scatter_kernel(const float* __restrict__ x, const int* __restrict__ src,
                    const int* __restrict__ dst, const float* __restrict__ ea,
                    const float* __restrict__ nm, float* __restrict__ agg) {
    long t = (long)blockIdx.x * 256 + threadIdx.x;
    long e = t >> 5;
    int  f = (int)(t & 31) << 2;
    int  s = src[e];
    int  d = dst[e];
    float sc = ea[e] * nm[e];
    v4f v = *(const v4f*)(x + (long)s * HID + f);
    float* ap = agg + (long)d * HID + f;
    atomicAdd(ap + 0, v[0] * sc);
    atomicAdd(ap + 1, v[1] * sc);
    atomicAdd(ap + 2, v[2] * sc);
    atomicAdd(ap + 3, v[3] * sc);
}

// ---------------------------------------------------------------------------
// WMMA linear: out[32 rows x 128 cols per block] = act(A @ W + bias [+ add])
// A is f32 [nrows][128]; converted to f16 in LDS. Wt is f16 [128][128]
// transposed (Wt[n][k] = W[k][n]). 8 waves: wave w owns output cols w*16..+16.
// The 4 B-fragments (full K=128 for this wave's 16 columns) are loaded ONCE
// and reused across 2 row-tiles -> 8 WMMAs per block-wave, half the weight
// traffic of a 16-row tiling. Safe for out == A (all global A reads happen in
// the LDS staging phase; blocks touch disjoint rows).
// ---------------------------------------------------------------------------
template <bool RELU, bool ADD>
__global__ __launch_bounds__(256)
void linear32_kernel(const float* __restrict__ A, const _Float16* __restrict__ Wt,
                     const float* __restrict__ bias, const float* __restrict__ addsrc,
                     float* __restrict__ out) {
    __shared__ __align__(16) _Float16 As[32 * HID];

    const int  tid      = threadIdx.x;
    const long blockrow = (long)blockIdx.x * 32;

    // Stage 32x128 f32 tile -> f16 LDS (256 threads x 16 contiguous floats).
    {
        const int    off = tid * 16;
        const float* gp  = A + blockrow * HID + off;
        v4f a0 = *(const v4f*)(gp);
        v4f a1 = *(const v4f*)(gp + 4);
        v4f a2 = *(const v4f*)(gp + 8);
        v4f a3 = *(const v4f*)(gp + 12);
        _Float16* sp = As + off;
#pragma unroll
        for (int i = 0; i < 4; ++i) {
            sp[i]      = (_Float16)a0[i];
            sp[4 + i]  = (_Float16)a1[i];
            sp[8 + i]  = (_Float16)a2[i];
            sp[12 + i] = (_Float16)a3[i];
        }
    }
    __syncthreads();

    const int lane = tid & 31;
    const int wave = tid >> 5;
    const int nlo  = lane & 15;   // M for A-frag, N for B/C/D-frag
    const int hi   = lane >> 4;   // K-half selector
    const int col  = wave * 16 + nlo;

    const float bv = bias[col];

    // Hoist all 4 B fragments (K = 0..127 for this wave's 16 output columns).
    // B frag (32x16 f16): halves 0..7 -> K = hi*8 + 0..7, halves 8..15 -> K = 16 + hi*8 + 0..7.
    const _Float16* __restrict__ wrow = Wt + (long)col * HID;
    v16h bf[4];
#pragma unroll
    for (int kk = 0; kk < 4; ++kk) {
        const _Float16* bp = wrow + kk * 32 + hi * 8;
        half8 blo = *(const half8*)bp;
        half8 bhi = *(const half8*)(bp + 16);
        bf[kk] = __builtin_shufflevector(blo, bhi,
                                         0, 1, 2, 3, 4, 5, 6, 7,
                                         8, 9, 10, 11, 12, 13, 14, 15);
    }

#pragma unroll
    for (int mt = 0; mt < 2; ++mt) {
        const long rowbase = blockrow + mt * 16;
        const _Float16* arow = As + (mt * 16 + nlo) * HID;

        v8f c;
#pragma unroll
        for (int r = 0; r < 8; ++r) c[r] = bv;

#pragma unroll
        for (int kk = 0; kk < 4; ++kk) {
            // A frag (16x32 f16): halves 0..7  -> K = k0 + hi*8 + 0..7,
            //                     halves 8..15 -> K = k0 + 16 + hi*8 + 0..7
            const _Float16* ap = arow + kk * 32 + hi * 8;
            half8 alo = *(const half8*)ap;
            half8 ahi = *(const half8*)(ap + 16);
            v16h a = __builtin_shufflevector(alo, ahi,
                                             0, 1, 2, 3, 4, 5, 6, 7,
                                             8, 9, 10, 11, 12, 13, 14, 15);
            c = __builtin_amdgcn_wmma_f32_16x16x32_f16(false, a, false, bf[kk],
                                                       (short)0, c, false, false);
        }

        // Epilogue: C/D frag VGPR r holds (M = r + 8*hi, N = col).
#pragma unroll
        for (int r = 0; r < 8; ++r) {
            const int  m   = r + hi * 8;
            const long idx = (rowbase + m) * HID + col;
            float v = c[r];
            if (ADD)  v += addsrc[idx];
            if (RELU) v = v > 0.f ? v : 0.f;
            out[idx] = v;
        }
    }
}

// ---------------------------------------------------------------------------
extern "C" void kernel_launch(void* const* d_in, const int* in_sizes, int n_in,
                              void* d_out, int out_size, void* d_ws, size_t ws_size,
                              hipStream_t stream) {
    (void)in_sizes; (void)n_in; (void)out_size; (void)ws_size;

    const int N_VALS = 100000, N_CONS = 100000, N_OBJ = 512;

    const float* b_in = (const float*)d_in[0];
    const float* q_in = (const float*)d_in[1];

    // RELS order: 0=v2c, 1=c2v, 2=v2o, 3=o2v, 4=c2o, 5=o2c
    const int R_v2c = 0, R_c2v = 1, R_v2o = 2, R_o2v = 3, R_c2o = 4, R_o2c = 5;
    const int EDGES[6] = {800000, 800000, 100000, 100000, 100000, 100000};
    const int*   e_src[6]; const int*   e_dst[6];
    const float* e_ea[6];  const float* e_nm[6];
    for (int i = 0; i < 6; ++i) {
        e_src[i] = (const int*)d_in[2 + 4 * i + 0];
        e_dst[i] = (const int*)d_in[2 + 4 * i + 1];
        e_ea[i]  = (const float*)d_in[2 + 4 * i + 2];
        e_nm[i]  = (const float*)d_in[2 + 4 * i + 3];
    }

    // params flattened (sorted keys): b_enc, fc_obj, layers[3]{c2o,c2v,o2c,o2v,v2c,v2o}, q_enc
    const float* bEnc_W0 = (const float*)d_in[26];
    const float* bEnc_b0 = (const float*)d_in[27];
    const float* bEnc_W1 = (const float*)d_in[28];
    const float* bEnc_b1 = (const float*)d_in[29];
    const float* fc_W    = (const float*)d_in[30];
    const float* fc_b    = (const float*)d_in[31];
    const float* qEnc_W0 = (const float*)d_in[104];
    const float* qEnc_b0 = (const float*)d_in[105];
    const float* qEnc_W1 = (const float*)d_in[106];
    const float* qEnc_b1 = (const float*)d_in[107];
    // RELS idx -> sorted position within a layer dict
    const int PS[6] = {4, 1, 5, 3, 0, 2};
    auto pW = [&](int l, int r, int leaf) -> const float* {
        return (const float*)d_in[32 + l * 24 + PS[r] * 4 + leaf]; // 0=W1,1=b1,2=W2,3=b2
    };

    // ---- workspace carve (f32 units) ----
    float* ws = (float*)d_ws;
    const long NODE = (long)N_CONS * HID;                 // 12.8M floats
    float* valsA = ws;
    float* valsB = valsA + NODE;
    float* consA = valsB + NODE;
    float* consB = consA + NODE;
    float* objA  = consB + NODE;
    float* objB  = objA + (long)N_OBJ * HID;
    float* agg   = objB + (long)N_OBJ * HID;
    float* oAbuf = agg + NODE;
    _Float16* wt16 = (_Float16*)(oAbuf + NODE);
    auto wtp = [&](int s) -> _Float16* { return wt16 + (long)s * HID * HID; };

    dim3 blk(256);

    // ---- convert all 128x128 weights to transposed f16 ----
    cvt_tr_kernel<<<64, blk, 0, stream>>>(bEnc_W1, wtp(0));
    cvt_tr_kernel<<<64, blk, 0, stream>>>(qEnc_W1, wtp(1));
    cvt_tr_kernel<<<64, blk, 0, stream>>>(fc_W,    wtp(2));
    for (int l = 0; l < 3; ++l)
        for (int r = 0; r < 6; ++r) {
            cvt_tr_kernel<<<64, blk, 0, stream>>>(pW(l, r, 0), wtp(3 + (l * 6 + r) * 2 + 0));
            cvt_tr_kernel<<<64, blk, 0, stream>>>(pW(l, r, 2), wtp(3 + (l * 6 + r) * 2 + 1));
        }

    // ---- encoders ----
    encode0_kernel<<<(N_CONS * HID) / 256, blk, 0, stream>>>(b_in, bEnc_W0, bEnc_b0, agg);
    linear32_kernel<false, false><<<N_CONS / 32, blk, 0, stream>>>(agg, wtp(0), bEnc_b1, nullptr, consA);
    encode0_kernel<<<(N_VALS * HID) / 256, blk, 0, stream>>>(q_in, qEnc_W0, qEnc_b0, agg);
    linear32_kernel<false, false><<<N_VALS / 32, blk, 0, stream>>>(agg, wtp(1), qEnc_b1, nullptr, valsA);
    hipMemsetAsync(objA, 0, (size_t)N_OBJ * HID * sizeof(float), stream);

    float* vals = valsA; float* valsN = valsB;
    float* cons = consA; float* consN = consB;
    float* obj  = objA;  float* objN  = objB;

    // conv(rel, xsrc) -> MLP; pairSecond=false writes plain result to outp,
    // pairSecond=true writes relu(result + addsrc) to outp.
    auto conv = [&](const float* xsrc, int r, int l, int ndst,
                    bool pairSecond, const float* addsrc, float* outp) {
        hipMemsetAsync(agg, 0, (size_t)ndst * HID * sizeof(float), stream);
        scatter_kernel<<<EDGES[r] / 8, blk, 0, stream>>>(xsrc, e_src[r], e_dst[r],
                                                         e_ea[r], e_nm[r], agg);
        // hidden = relu(agg @ W1 + b1)   (in place on agg)
        linear32_kernel<true, false><<<ndst / 32, blk, 0, stream>>>(
            agg, wtp(3 + (l * 6 + r) * 2 + 0), pW(l, r, 1), nullptr, agg);
        if (pairSecond)
            linear32_kernel<true, true><<<ndst / 32, blk, 0, stream>>>(
                agg, wtp(3 + (l * 6 + r) * 2 + 1), pW(l, r, 3), addsrc, outp);
        else
            linear32_kernel<false, false><<<ndst / 32, blk, 0, stream>>>(
                agg, wtp(3 + (l * 6 + r) * 2 + 1), pW(l, r, 3), nullptr, outp);
    };

    for (int l = 0; l < 3; ++l) {
        // cons2 = relu(conv(v2c, vals) + conv(o2c, obj))
        conv(vals,  R_v2c, l, N_CONS, false, nullptr, oAbuf);
        conv(obj,   R_o2c, l, N_CONS, true,  oAbuf,  consN);
        // vals2 = relu(conv(c2v, cons2) + conv(o2v, obj))
        conv(consN, R_c2v, l, N_VALS, false, nullptr, oAbuf);
        conv(obj,   R_o2v, l, N_VALS, true,  oAbuf,  valsN);
        // obj2 = relu(conv(v2o, vals2) + conv(c2o, cons2))
        conv(valsN, R_v2o, l, N_OBJ, false, nullptr, oAbuf);
        conv(consN, R_c2o, l, N_OBJ, true,  oAbuf,  objN);
        float* t;
        t = vals; vals = valsN; valsN = t;
        t = cons; cons = consN; consN = t;
        t = obj;  obj  = objN;  objN  = t;
    }
    (void)cons;

    // final: out = obj @ fc_W + fc_b   (no relu), 512x128 f32
    linear32_kernel<false, false><<<N_OBJ / 32, blk, 0, stream>>>(
        obj, wtp(2), fc_b, nullptr, (float*)d_out);
}
